// RainFault_33371895890245
// MI455X (gfx1250) — 8-bit matrix operand order repaired and checked by reading, compiled, not run on hardware
//
#include <hip/hip_runtime.h>
#include <hip/hip_bf16.h>

typedef __attribute__((ext_vector_type(16))) int   v16i;
typedef __attribute__((ext_vector_type(8)))  float v8f;

#define Bv 32
#define Cv 3
#define Hv 512
#define Wv 512
#define NS 100

__global__ __launch_bounds__(256)
void rainfault_wmma_kernel(const float* __restrict__ x,
                           const int*   __restrict__ ys,
                           const int*   __restrict__ xcs,
                           const int*   __restrict__ lens,
                           float*       __restrict__ out)
{
    __shared__ unsigned hmask[32][4];
    __shared__ unsigned wmask[128][4];

    const int tid = threadIdx.x;
    const int b   = blockIdx.z;
    const int h0  = blockIdx.y << 5;
    const int w0b = blockIdx.x << 7;

    unsigned* wflat = &wmask[0][0];
    unsigned* hflat = &hmask[0][0];
    wflat[tid]       = 0u;
    wflat[tid + 256] = 0u;
    if (tid < 128) hflat[tid] = 0u;
    __syncthreads();

    if (tid < NS) {
        const int s   = tid;
        const int y   = ys  [b * NS + s];
        const int xc  = xcs [b * NS + s];
        const int len = lens[b * NS + s];
        const unsigned bit = 1u << (s & 31);
        const int word = s >> 5;

        int lo = y       > h0      ? y       : h0;
        int hi = y + len < h0 + 32 ? y + len : h0 + 32;
        for (int h = lo; h < hi; ++h)
            atomicOr(&hmask[h - h0][word], bit);

        #pragma unroll
        for (int d = 0; d < 2; ++d) {
            const int xx = xc - d;
            if (xx >= w0b && xx < w0b + 128)
                atomicOr(&wmask[xx - w0b][word], bit);
        }
    }
    __syncthreads();

    const int lane = tid & 31;
    const int wave = tid >> 5;
    const int g    = lane >> 4;
    const int MN   = lane & 15;
    const int wl   = (wave << 4) + MN;
    const int w    = w0b + wl;

    v16i bm;
    #pragma unroll
    for (int v = 0; v < 16; ++v) {
        const int kb = 64 * (v >> 3) + 16 * ((v >> 1) & 3) + 8 * g + 4 * (v & 1);
        const unsigned nb = (wmask[wl][kb >> 5] >> (kb & 31)) & 0xFu;
        bm[v] = (int)(((nb * 0x00204081u) & 0x01010101u) * 0x38u);
    }

    #pragma unroll
    for (int t = 0; t < 2; ++t) {
        const int hbase = h0 + (t << 4);

        v16i a;
        #pragma unroll
        for (int v = 0; v < 16; ++v) {
            const int ka = 64 * (v >> 3) + 16 * ((v >> 1) & 3) + 8 * g + 4 * (v & 1);
            const unsigned na = (hmask[(t << 4) + MN][ka >> 5] >> (ka & 31)) & 0xFu;
            a[v] = (int)(((na * 0x00204081u) & 0x01010101u) * 0x38u);
        }

        v8f c = {0.f, 0.f, 0.f, 0.f, 0.f, 0.f, 0.f, 0.f};
        c = __builtin_amdgcn_wmma_f32_16x16x128_fp8_fp8(a, bm, (short)0, c,
                                                         false,
                                                         false);

        #pragma unroll
        for (int r = 0; r < 8; ++r) {
            const int   h      = hbase + r + 8 * g;
            const int   cnt    = (int)(c[r] + 0.5f);
            const float factor = ldexpf(1.0f, -cnt);
            const float addend = 1.0f - factor;
            #pragma unroll
            for (int ch = 0; ch < 3; ++ch) {
                const size_t idx = (((size_t)b * Cv + ch) * Hv + h) * Wv + w;
                float vv = fmaf(x[idx], factor, addend);
                vv = fminf(fmaxf(vv, 0.0f), 1.0f);
                out[idx] = vv;
            }
        }
    }
}

extern "C" void kernel_launch(void* const* d_in, const int* in_sizes, int n_in,
                              void* d_out, int out_size, void* d_ws, size_t ws_size,
                              hipStream_t stream)
{
    const float* x    = (const float*)d_in[0];
    const int*   ys   = (const int*)  d_in[1];
    const int*   xcs  = (const int*)  d_in[2];
    const int*   lens = (const int*)  d_in[3];
    float*       out  = (float*)      d_out;

    dim3 grid(Wv / 128, Hv / 32, Bv);
    rainfault_wmma_kernel<<<grid, 256, 0, stream>>>(x, ys, xcs, lens, out);
}
